// MultiHeadAttention_62113817034808
// MI455X (gfx1250) — matrix operand order repaired and checked by reading, not compiled, not run on hardware
//
#include <hip/hip_runtime.h>

#define B_SZ 4
#define SEQ  2048
#define DM   1024
#define NH   16
#define DK   64

#if defined(__has_builtin)
#  if __has_builtin(__builtin_amdgcn_global_load_async_to_lds_b128)
#    define ATTN_ASYNC 1
#  endif
#endif
#ifndef ATTN_ASYNC
#  define ATTN_ASYNC 0
#endif

typedef __bf16 v16bf __attribute__((ext_vector_type(16)));
typedef float  v8f   __attribute__((ext_vector_type(8)));
typedef int    i32x4 __attribute__((vector_size(16)));

union Frag {
    v16bf          v;
    __bf16         h[16];
    unsigned int   u[8];
    uint4          q[2];
};

// Native f32 -> bf16 convert (codegen picks v_cvt_pk_bf16_f32).
__device__ __forceinline__ unsigned short bf16u(float a) {
    union { __bf16 h; unsigned short s; } u;
    u.h = (__bf16)a;
    return u.s;
}

__device__ __forceinline__ v8f wmma_bf16(const Frag& a, const Frag& b, v8f c) {
    return __builtin_amdgcn_wmma_f32_16x16x32_bf16(
        false, a.v, false, b.v, (short)0, c, false, false);
}

#if ATTN_ASYNC
__device__ __forceinline__ void async_cp_b128(const unsigned short* g, unsigned short* l) {
    __builtin_amdgcn_global_load_async_to_lds_b128(
        (__attribute__((address_space(1))) i32x4*)(g),
        (__attribute__((address_space(3))) i32x4*)(l), 0, 0);
}
__device__ __forceinline__ void wait_async_le4() {
#  if __has_builtin(__builtin_amdgcn_s_wait_asynccnt)
    __builtin_amdgcn_s_wait_asynccnt(4);
#  else
    asm volatile("s_wait_asynccnt 0x4" ::: "memory");
#  endif
}
__device__ __forceinline__ void wait_async_le0() {
#  if __has_builtin(__builtin_amdgcn_s_wait_asynccnt)
    __builtin_amdgcn_s_wait_asynccnt(0);
#  else
    asm volatile("s_wait_asynccnt 0x0" ::: "memory");
#  endif
}
#endif

// ---------------------------------------------------------------------------
// Kernel 1: per-head QKV projections.  X[b,s,h*64+d] (f32) @ W[h,d,e] + bias.
// Writes q,k row-major bf16 [bh, s, dk]; v transposed bf16 [bh, dk, s].
// ---------------------------------------------------------------------------
__global__ __launch_bounds__(128) void proj_qkv(
    const float* __restrict__ Q, const float* __restrict__ K, const float* __restrict__ V,
    const float* __restrict__ Wq, const float* __restrict__ bq,
    const float* __restrict__ Wk, const float* __restrict__ bk,
    const float* __restrict__ Wv, const float* __restrict__ bv,
    unsigned short* __restrict__ q_ws, unsigned short* __restrict__ k_ws,
    unsigned short* __restrict__ vT_ws)
{
    __shared__ unsigned short sWT[3 * DK * DK];  // W^T in bf16 per head, 24 KB

    const int tid = threadIdx.x;
    const int h = blockIdx.y, b = blockIdx.z;
    const int s0 = blockIdx.x * 128;

    const float* Ws[3] = { Wq + h * DK * DK, Wk + h * DK * DK, Wv + h * DK * DK };
    for (int idx = tid; idx < 3 * DK * DK; idx += 128) {
        int m = idx >> 12;            // which matrix (4096 each)
        int r = idx & 4095;
        int d = r >> 6, e = r & 63;
        sWT[(m << 12) + e * DK + d] = bf16u(Ws[m][r]);
    }
    __syncthreads();

    const int wave = tid >> 5, lane = tid & 31, lo = lane & 15, hi = lane >> 4;
    const float* Xs[3] = { Q, K, V };
    const float* Bs[3] = { bq + h * DK, bk + h * DK, bv + h * DK };

#pragma unroll
    for (int mat = 0; mat < 3; ++mat) {        // unrolled: store select is static
        const float* X = Xs[mat];
        const unsigned short* WT = sWT + (mat << 12);
        for (int mt = 0; mt < 2; ++mt) {
            const int row0 = s0 + wave * 32 + mt * 16;
            v8f zero = {0.f,0.f,0.f,0.f,0.f,0.f,0.f,0.f};
            v8f acc[4] = { zero, zero, zero, zero };

#pragma unroll
            for (int ks = 0; ks < 2; ++ks) {
                const int kk0 = ks * 32;
                const float* rp = X + ((size_t)b * SEQ + row0 + lo) * DM + h * DK + kk0;
                Frag A;
                float4 f0 = *(const float4*)(rp + 8 * hi);
                float4 f1 = *(const float4*)(rp + 8 * hi + 4);
                float4 g0 = *(const float4*)(rp + 16 + 8 * hi);
                float4 g1 = *(const float4*)(rp + 16 + 8 * hi + 4);
                A.h[0]  = (__bf16)f0.x; A.h[1]  = (__bf16)f0.y;
                A.h[2]  = (__bf16)f0.z; A.h[3]  = (__bf16)f0.w;
                A.h[4]  = (__bf16)f1.x; A.h[5]  = (__bf16)f1.y;
                A.h[6]  = (__bf16)f1.z; A.h[7]  = (__bf16)f1.w;
                A.h[8]  = (__bf16)g0.x; A.h[9]  = (__bf16)g0.y;
                A.h[10] = (__bf16)g0.z; A.h[11] = (__bf16)g0.w;
                A.h[12] = (__bf16)g1.x; A.h[13] = (__bf16)g1.y;
                A.h[14] = (__bf16)g1.z; A.h[15] = (__bf16)g1.w;
#pragma unroll
                for (int nt = 0; nt < 4; ++nt) {
                    Frag Bw;
                    const unsigned short* wp = WT + (nt * 16 + lo) * DK + kk0 + 8 * hi;
                    Bw.q[0] = *(const uint4*)wp;
                    Bw.q[1] = *(const uint4*)(wp + 16);
                    acc[nt] = wmma_bf16(A, Bw, acc[nt]);
                }
            }
#pragma unroll
            for (int nt = 0; nt < 4; ++nt) {
                const float bias = Bs[mat][nt * 16 + lo];
#pragma unroll
                for (int i = 0; i < 8; ++i) {
                    const float val = acc[nt][i] + bias;
                    const int row = row0 + i + 8 * hi;
                    const int e = nt * 16 + lo;
                    if (mat == 0)
                        q_ws[((size_t)(b * NH + h) * SEQ + row) * DK + e] = bf16u(val);
                    else if (mat == 1)
                        k_ws[((size_t)(b * NH + h) * SEQ + row) * DK + e] = bf16u(val);
                    else
                        vT_ws[((size_t)(b * NH + h) * DK + e) * SEQ + row] = bf16u(val);
                }
            }
        }
    }
}

// ---------------------------------------------------------------------------
// Kernel 2: Wo (f32, [d,e]) -> Wo^T (bf16, [e,d]) so B-fragments are contiguous.
// ---------------------------------------------------------------------------
__global__ __launch_bounds__(256) void cvt_wo(const float* __restrict__ Wo,
                                              unsigned short* __restrict__ wot)
{
    const int idx = blockIdx.x * 256 + threadIdx.x;
    if (idx < DM * DM) {
        const int e = idx & (DM - 1);
        const int d = idx >> 10;
        wot[(size_t)e * DM + d] = bf16u(Wo[idx]);
    }
}

// ---------------------------------------------------------------------------
// Kernel 3: flash attention per (b,h).  Block = 4 waves x 16 query rows.
// 64-key chunks, online softmax, P re-layout through LDS, ctx via WMMA vs v^T.
// When the async-to-LDS builtin exists, the K chunk is staged into LDS once
// per block (double-buffered, ASYNCcnt-tracked) instead of 4x from global.
// ---------------------------------------------------------------------------
__global__ __launch_bounds__(128) void attn(
    const unsigned short* __restrict__ q_ws, const unsigned short* __restrict__ k_ws,
    const unsigned short* __restrict__ vT_ws, unsigned short* __restrict__ ctx_ws)
{
    __shared__ unsigned short sP[4 * 16 * 64];      // per-wave P tiles, 8 KB
#if ATTN_ASYNC
    __shared__ unsigned short sK[2][64 * DK];       // K chunk double buffer, 16 KB
#endif

    const int tid = threadIdx.x, wave = tid >> 5, lane = tid & 31;
    const int lo = lane & 15, hi = lane >> 4;
    const int h = blockIdx.y, b = blockIdx.z;
    const int q0 = blockIdx.x * 64 + wave * 16;

    const unsigned short* qb  = q_ws  + (size_t)(b * NH + h) * SEQ * DK;
    const unsigned short* kb  = k_ws  + (size_t)(b * NH + h) * SEQ * DK;
    const unsigned short* vtb = vT_ws + (size_t)(b * NH + h) * DK * SEQ;

    Frag Aq[2];
#pragma unroll
    for (int ks = 0; ks < 2; ++ks) {
        const unsigned short* qp = qb + (size_t)(q0 + lo) * DK + ks * 32;
        Aq[ks].q[0] = *(const uint4*)(qp + 16 * hi);
        Aq[ks].q[1] = *(const uint4*)(qp + 16 * hi + 8);
    }

    v8f zero = {0.f,0.f,0.f,0.f,0.f,0.f,0.f,0.f};
    v8f acc[4] = { zero, zero, zero, zero };
    float mr[8], lr[8];
#pragma unroll
    for (int i = 0; i < 8; ++i) { mr[i] = -3.0e38f; lr[i] = 0.f; }

    unsigned short* pbase = sP + wave * 16 * 64;

#if ATTN_ASYNC
    // stage chunk 0: 64 rows x 128 B = 8 KB; 4 x b128 per thread
    {
        const unsigned short* g = kb + tid * 32;
        unsigned short* l = &sK[0][tid * 32];
#pragma unroll
        for (int j = 0; j < 4; ++j) async_cp_b128(g + j * 8, l + j * 8);
    }
#endif

    for (int kc = 0; kc < SEQ; kc += 64) {
#if ATTN_ASYNC
        const int cur = (kc >> 6) & 1;
        if (kc + 64 < SEQ) {   // kick off next chunk, then drain current (in-order)
            const unsigned short* g = kb + (size_t)(kc + 64) * DK + tid * 32;
            unsigned short* l = &sK[cur ^ 1][tid * 32];
#pragma unroll
            for (int j = 0; j < 4; ++j) async_cp_b128(g + j * 8, l + j * 8);
            wait_async_le4();
        } else {
            wait_async_le0();
        }
        __syncthreads();                       // chunk visible to all 4 waves
        const unsigned short* kchunk = &sK[cur][0];
#endif
        // prefetch next v^T span while we compute on this chunk
        if (kc + 64 < SEQ)
            __builtin_prefetch(vtb + (size_t)(lane * 2) * SEQ + kc + 64, 0, 1);

        // scores tile: 16 queries x 64 keys
        v8f st[4];
#pragma unroll
        for (int nt = 0; nt < 4; ++nt) {
            v8f c = zero;
#pragma unroll
            for (int ks = 0; ks < 2; ++ks) {
                Frag Bk;
#if ATTN_ASYNC
                const unsigned short* kp =
                    kchunk + (nt * 16 + lo) * DK + ks * 32 + 16 * hi;
#else
                const unsigned short* kp =
                    kb + (size_t)(kc + nt * 16 + lo) * DK + ks * 32 + 16 * hi;
#endif
                Bk.q[0] = *(const uint4*)kp;
                Bk.q[1] = *(const uint4*)(kp + 8);
                c = wmma_bf16(Aq[ks], Bk, c);
            }
#pragma unroll
            for (int i = 0; i < 8; ++i) c[i] *= 0.125f;   // 1/sqrt(64)
            st[nt] = c;
        }
        // online softmax: row max / correction / exp / row sum
        float corr[8];
#pragma unroll
        for (int i = 0; i < 8; ++i) {
            float t = fmaxf(fmaxf(st[0][i], st[1][i]), fmaxf(st[2][i], st[3][i]));
            t = fmaxf(t, __shfl_xor(t, 1, 32));
            t = fmaxf(t, __shfl_xor(t, 2, 32));
            t = fmaxf(t, __shfl_xor(t, 4, 32));
            t = fmaxf(t, __shfl_xor(t, 8, 32));
            const float nm = fmaxf(mr[i], t);
            corr[i] = __expf(mr[i] - nm);
            mr[i] = nm;
        }
#pragma unroll
        for (int nt = 0; nt < 4; ++nt)
#pragma unroll
            for (int i = 0; i < 8; ++i)
                st[nt][i] = __expf(st[nt][i] - mr[i]);
#pragma unroll
        for (int i = 0; i < 8; ++i) {
            float s = st[0][i] + st[1][i] + st[2][i] + st[3][i];
            s += __shfl_xor(s, 1, 32);
            s += __shfl_xor(s, 2, 32);
            s += __shfl_xor(s, 4, 32);
            s += __shfl_xor(s, 8, 32);
            lr[i] = lr[i] * corr[i] + s;
        }
#pragma unroll
        for (int nt = 0; nt < 4; ++nt)
#pragma unroll
            for (int i = 0; i < 8; ++i)
                acc[nt][i] *= corr[i];

        // C-layout P -> LDS -> A-layout P (bf16)
#pragma unroll
        for (int nt = 0; nt < 4; ++nt)
#pragma unroll
            for (int i = 0; i < 8; ++i)
                pbase[(i + 8 * hi) * 64 + nt * 16 + lo] = bf16u(st[nt][i]);
        __syncthreads();

        Frag Ap[2];
#pragma unroll
        for (int ks = 0; ks < 2; ++ks) {
            const unsigned short* pp = pbase + lo * 64 + ks * 32;
            Ap[ks].q[0] = *(const uint4*)(pp + 16 * hi);
            Ap[ks].q[1] = *(const uint4*)(pp + 16 * hi + 8);
        }
        // ctx += P @ V   (B-fragments contiguous thanks to v^T layout)
#pragma unroll
        for (int nt = 0; nt < 4; ++nt) {
#pragma unroll
            for (int ks = 0; ks < 2; ++ks) {
                Frag Bv;
                const unsigned short* vp =
                    vtb + (size_t)(nt * 16 + lo) * SEQ + kc + ks * 32 + 16 * hi;
                Bv.q[0] = *(const uint4*)vp;
                Bv.q[1] = *(const uint4*)(vp + 8);
                acc[nt] = wmma_bf16(Ap[ks], Bv, acc[nt]);
            }
        }
        __syncthreads();   // also fences K buffer reuse two iterations out
    }

#pragma unroll
    for (int i = 0; i < 8; ++i) lr[i] = 1.0f / lr[i];
#pragma unroll
    for (int nt = 0; nt < 4; ++nt)
#pragma unroll
        for (int i = 0; i < 8; ++i) {
            const float val = acc[nt][i] * lr[i];
            const size_t row = (size_t)b * SEQ + q0 + i + 8 * hi;
            ctx_ws[row * DM + h * DK + nt * 16 + lo] = bf16u(val);
        }
}

// ---------------------------------------------------------------------------
// Kernel 4: output projection  out = ctx(bf16) @ Wo + bo  -> f32.
// Block = 4 waves; wave = 16 rows x 64 cols tile; K loop over 1024.
// ---------------------------------------------------------------------------
__global__ __launch_bounds__(128) void out_proj(
    const unsigned short* __restrict__ ctx_ws, const unsigned short* __restrict__ wot,
    const float* __restrict__ bo, float* __restrict__ out)
{
    const int tid = threadIdx.x, wave = tid >> 5, lane = tid & 31;
    const int lo = lane & 15, hi = lane >> 4;
    const int r0 = blockIdx.x * 64 + wave * 16;
    const int c0 = blockIdx.y * 64;

    v8f zero = {0.f,0.f,0.f,0.f,0.f,0.f,0.f,0.f};
    v8f acc[4] = { zero, zero, zero, zero };

    for (int kk0 = 0; kk0 < DM; kk0 += 32) {
        Frag A;
        const unsigned short* ap = ctx_ws + (size_t)(r0 + lo) * DM + kk0;
        A.q[0] = *(const uint4*)(ap + 16 * hi);
        A.q[1] = *(const uint4*)(ap + 16 * hi + 8);
#pragma unroll
        for (int nt = 0; nt < 4; ++nt) {
            Frag Bw;
            const unsigned short* wp = wot + (size_t)(c0 + nt * 16 + lo) * DM + kk0 + 16 * hi;
            Bw.q[0] = *(const uint4*)wp;
            Bw.q[1] = *(const uint4*)(wp + 8);
            acc[nt] = wmma_bf16(A, Bw, acc[nt]);
        }
    }
#pragma unroll
    for (int nt = 0; nt < 4; ++nt) {
        const float bias = bo[c0 + nt * 16 + lo];
#pragma unroll
        for (int i = 0; i < 8; ++i)
            out[(size_t)(r0 + i + 8 * hi) * DM + c0 + nt * 16 + lo] = acc[nt][i] + bias;
    }
}

// ---------------------------------------------------------------------------
extern "C" void kernel_launch(void* const* d_in, const int* in_sizes, int n_in,
                              void* d_out, int out_size, void* d_ws, size_t ws_size,
                              hipStream_t stream)
{
    (void)in_sizes; (void)n_in; (void)out_size; (void)ws_size;

    const float* Q  = (const float*)d_in[0];
    const float* K  = (const float*)d_in[1];
    const float* V  = (const float*)d_in[2];
    const float* Wq = (const float*)d_in[3];
    const float* bq = (const float*)d_in[4];
    const float* Wk = (const float*)d_in[5];
    const float* bk = (const float*)d_in[6];
    const float* Wv = (const float*)d_in[7];
    const float* bv = (const float*)d_in[8];
    const float* Wo = (const float*)d_in[9];
    const float* bo = (const float*)d_in[10];
    float* out = (float*)d_out;

    const size_t NQKV = (size_t)B_SZ * NH * SEQ * DK;   // 8,388,608 elems
    unsigned short* q_ws   = (unsigned short*)d_ws;
    unsigned short* k_ws   = q_ws + NQKV;
    unsigned short* vT_ws  = k_ws + NQKV;
    unsigned short* ctx_ws = vT_ws + NQKV;
    unsigned short* wot    = ctx_ws + (size_t)B_SZ * SEQ * DM;

    proj_qkv<<<dim3(SEQ / 128, NH, B_SZ), 128, 0, stream>>>(
        Q, K, V, Wq, bq, Wk, bk, Wv, bv, q_ws, k_ws, vT_ws);
    cvt_wo<<<dim3((DM * DM) / 256), 256, 0, stream>>>(Wo, wot);
    attn<<<dim3(SEQ / 64, NH, B_SZ), 128, 0, stream>>>(q_ws, k_ws, vT_ws, ctx_ws);
    out_proj<<<dim3((B_SZ * SEQ) / 64, DM / 64), 128, 0, stream>>>(ctx_ws, wot, bo, out);
}
